// EdgeEarlyInteraction_45140106281536
// MI455X (gfx1250) — compile-verified, matrix operand order checked by reading, not run on hardware
//
#include <hip/hip_runtime.h>
#include <hip/hip_bf16.h>
#include <stdint.h>

typedef _Float16 half_t;
typedef __attribute__((ext_vector_type(16))) _Float16 v16h;
typedef __attribute__((ext_vector_type(8)))  _Float16 v8h;
typedef __attribute__((ext_vector_type(8)))  float    v8f;
typedef __attribute__((ext_vector_type(4)))  float    v4f;

#define CDIV(a,b) (((a)+(b)-1)/(b))
#define CAT16(lo,hi) __builtin_shufflevector(lo, hi, 0,1,2,3,4,5,6,7,8,9,10,11,12,13,14,15)

// ---------------------------------------------------------------------------
// Elementwise utility kernels (vector forms require extent % 8 == 0, 16B align)
// ---------------------------------------------------------------------------
__global__ void k_cvt(half_t* __restrict__ d, const float* __restrict__ s, long n) {
  long i = (long)blockIdx.x * blockDim.x + threadIdx.x;
  if (i < n) d[i] = (half_t)s[i];
}
// transpose-convert: s is Krows x Ncols f32 row-major; d is Ncols x Krows f16
__global__ void k_cvt_t(half_t* __restrict__ d, const float* __restrict__ s,
                        int Ncols, int Krows) {
  long i = (long)blockIdx.x * blockDim.x + threadIdx.x;
  if (i >= (long)Ncols * Krows) return;
  int n = (int)(i / Krows), k = (int)(i % Krows);
  d[i] = (half_t)s[(long)k * Ncols + n];
}
// transpose-convert, dest row stride: d[n*dstride + k] = s[k*Ncols + n], k < Krows
__global__ void k_cvt_ts(half_t* __restrict__ d, const float* __restrict__ s,
                         int Ncols, int Krows, int dstride) {
  long i = (long)blockIdx.x * blockDim.x + threadIdx.x;
  if (i >= (long)Ncols * Krows) return;
  int n = (int)(i / Krows), k = (int)(i % Krows);
  d[(long)n * dstride + k] = (half_t)s[(long)k * Ncols + n];
}
// convert E x 16 f32 into E x 32 f16 (upper K half pre-zeroed)
__global__ void k_cvt_pad(half_t* __restrict__ d, const float* __restrict__ s, int E) {
  long i = (long)blockIdx.x * blockDim.x + threadIdx.x;
  if (i >= (long)E * 16) return;
  int e = (int)(i >> 4), c = (int)(i & 15);
  d[(long)e * 32 + c] = (half_t)s[i];
}
__global__ void k_copy_f32v(v4f* __restrict__ d, const v4f* __restrict__ s, long n4) {
  long i = (long)blockIdx.x * blockDim.x + threadIdx.x;
  if (i < n4) d[i] = s[i];
}
__global__ void k_copy_f16v(v8h* __restrict__ d, const v8h* __restrict__ s, long n8) {
  long i = (long)blockIdx.x * blockDim.x + threadIdx.x;
  if (i < n8) d[i] = s[i];
}
__global__ void k_zero_f32v(v4f* __restrict__ d, long n4) {
  long i = (long)blockIdx.x * blockDim.x + threadIdx.x;
  if (i < n4) d[i] = (v4f){0.f, 0.f, 0.f, 0.f};
}
__global__ void k_zero_f16v(v8h* __restrict__ d, long n8) {
  long i = (long)blockIdx.x * blockDim.x + threadIdx.x;
  if (i < n8) d[i] = (v8h)(_Float16)0.f;
}

// ---------------------------------------------------------------------------
// Batched WMMA GEMM:  C[z] = act(scale * A[z] x B[z] + bias)
//   A: M x K row-major f16; REQUIRES M % 32 == 0, K % 32 == 0, all leading
//   dims multiples of 8 halves (16B alignment for b128 fragment loads).
//   TB==1: B stored N x K (B^T), contiguous 16B loads. TB==0: B is K x N.
//   N % 64 == 0. Block = 128 threads = 4 waves; each wave owns a 32x64 strip:
//   2 A fragments x 4 B fragments -> 8 v_wmma per K-step, each B reused 2x.
// ---------------------------------------------------------------------------
template<int ACT, int TB>
__global__ void k_gemm(const half_t* __restrict__ A, const half_t* __restrict__ B,
                       const float* __restrict__ bias,
                       float* __restrict__ Cf, half_t* __restrict__ Ch,
                       int M, int N, int K, int lda, int ldb, int ldc,
                       long sA, long sB, long sC, float scale) {
  int wave = threadIdx.x >> 5;
  int lane = threadIdx.x & 31;
  int row0 = blockIdx.x * 32;
  int col0 = (blockIdx.y * 4 + wave) * 64;
  if (col0 >= N) return;                         // uniform per-wave exit
  long z = blockIdx.z;
  const half_t* Az = A + z * sA;
  const half_t* Bz = B + z * sB;

  int l15  = lane & 15;
  int kofs = (lane >> 4) << 3;
  const half_t* pa0 = Az + (long)(row0 + l15) * lda + kofs;
  const half_t* pa1 = pa0 + (long)16 * lda;

  v8f acc[4][2] = {};
  if (TB) {
    const half_t* pb[4];
#pragma unroll
    for (int t = 0; t < 4; ++t)
      pb[t] = Bz + (long)(col0 + t * 16 + l15) * ldb + kofs;
    for (int k0 = 0; k0 < K; k0 += 32) {
      v16h a0 = CAT16(*(const v8h*)(pa0 + k0), *(const v8h*)(pa0 + k0 + 16));
      v16h a1 = CAT16(*(const v8h*)(pa1 + k0), *(const v8h*)(pa1 + k0 + 16));
#pragma unroll
      for (int t = 0; t < 4; ++t) {
        v16h b = CAT16(*(const v8h*)(pb[t] + k0), *(const v8h*)(pb[t] + k0 + 16));
        acc[t][0] = __builtin_amdgcn_wmma_f32_16x16x32_f16(false, a0, false, b,
                                                           (short)0, acc[t][0], false, false);
        acc[t][1] = __builtin_amdgcn_wmma_f32_16x16x32_f16(false, a1, false, b,
                                                           (short)0, acc[t][1], false, false);
      }
    }
  } else {
    for (int k0 = 0; k0 < K; k0 += 32) {
      v16h a0 = CAT16(*(const v8h*)(pa0 + k0), *(const v8h*)(pa0 + k0 + 16));
      v16h a1 = CAT16(*(const v8h*)(pa1 + k0), *(const v8h*)(pa1 + k0 + 16));
      const half_t* prow = Bz + (long)(k0 + kofs) * ldb + col0 + l15;
#pragma unroll
      for (int t = 0; t < 4; ++t) {
        const half_t* pt = prow + t * 16;
        v16h b;
#pragma unroll
        for (int v = 0; v < 8; ++v) {
          int dk = (v < 4) ? (v << 1) : (16 + ((v - 4) << 1));
          b[2*v]   = pt[(long)dk * ldb];
          b[2*v+1] = pt[(long)(dk + 1) * ldb];
        }
        acc[t][0] = __builtin_amdgcn_wmma_f32_16x16x32_f16(false, a0, false, b,
                                                           (short)0, acc[t][0], false, false);
        acc[t][1] = __builtin_amdgcn_wmma_f32_16x16x32_f16(false, a1, false, b,
                                                           (short)0, acc[t][1], false, false);
      }
    }
  }

  int rbase = (lane >> 4) << 3;
#pragma unroll
  for (int t = 0; t < 4; ++t) {
    int col = col0 + t * 16 + l15;
    float bv = bias ? bias[col] : 0.f;
#pragma unroll
    for (int m = 0; m < 2; ++m) {
#pragma unroll
      for (int r = 0; r < 8; ++r) {
        int row = row0 + m * 16 + rbase + r;
        float v = acc[t][m][r] * scale + bv;
        if (ACT) v = fmaxf(v, 0.f);
        long idx = z * sC + (long)row * ldc + col;
        if (Cf) Cf[idx] = v;
        if (Ch) Ch[idx] = (half_t)v;
      }
    }
  }
}

// ---------------------------------------------------------------------------
// Graph-specific kernels
// ---------------------------------------------------------------------------
__global__ void k_build_min(half_t* __restrict__ m_in, const half_t* __restrict__ nodes,
                            const float* __restrict__ inter, int inter_ld,
                            const int* __restrict__ from_idx, const int* __restrict__ to_idx,
                            int E) {
  long i = (long)blockIdx.x * blockDim.x + threadIdx.x;
  if (i >= (long)E * 256) return;
  int e = (int)(i >> 8), c = (int)(i & 255);
  half_t v;
  if (c < 64)       v = nodes[(long)from_idx[e] * 64 + c];
  else if (c < 128) v = nodes[(long)to_idx[e]   * 64 + (c - 64)];
  else              v = (half_t)inter[(long)e * inter_ld + (c - 128)];
  m_in[i] = v;
}

__global__ void k_segsum(float* __restrict__ agg, const float* __restrict__ msg, int msg_ld,
                         const int* __restrict__ to_idx, int E) {
  long i = (long)blockIdx.x * blockDim.x + threadIdx.x;
  if (i >= (long)E * 128) return;
  int e = (int)(i >> 7), j = (int)(i & 127);
  atomicAdd(&agg[(long)to_idx[e] * 128 + j], msg[(long)e * msg_ld + j]);
}

__global__ void k_build_uin(half_t* __restrict__ u_in, const half_t* __restrict__ nodes,
                            const float* __restrict__ agg, int Nn) {
  long i = (long)blockIdx.x * blockDim.x + threadIdx.x;
  if (i >= (long)Nn * 192) return;
  int r = (int)(i / 192), c = (int)(i % 192);
  u_in[i] = (c < 64) ? nodes[(long)r * 64 + c] : (half_t)agg[(long)r * 128 + (c - 64)];
}

__global__ void k_scatter(half_t* __restrict__ flat, float* __restrict__ qfin,
                          float* __restrict__ cfin, const float* __restrict__ msg,
                          const int* __restrict__ edge_indices, int E) {
  long i = (long)blockIdx.x * blockDim.x + threadIdx.x;
  if (i >= (long)E * 640) return;
  int e = (int)(i / 640), j = (int)(i % 640);
  int slot = edge_indices[e];
  int g = slot >> 8, pos = slot & 255;
  float v = msg[(long)e * 640 + j];
  flat[(long)slot * 640 + j] = (half_t)v;
  if (j >= 512) {
    float* dst = (g & 1) ? cfin : qfin;
    dst[((long)(g >> 1) * 256 + pos) * 128 + (j - 512)] = v;
  }
}

__global__ void k_mask_rows(half_t* __restrict__ t, const int* __restrict__ sizes) {
  long i = (long)blockIdx.x * blockDim.x + threadIdx.x;
  if (i >= 64L * 256 * 64) return;
  int b = (int)(i >> 14);
  int row = (int)((i >> 6) & 255);
  if (row >= sizes[b]) t[i] = (half_t)0.f;
}

__global__ void k_sink_norm(float* __restrict__ la, int axis) {
  int b = blockIdx.x >> 8, r = blockIdx.x & 255;
  long base; int stride;
  if (axis == 0) { base = ((long)b * 256 + r) * 256; stride = 1; }
  else           { base = (long)b * 65536 + r;       stride = 256; }
  int t = threadIdx.x;
  float v = la[base + (long)t * stride];
  __shared__ float sm[256];
  sm[t] = v; __syncthreads();
  for (int s = 128; s > 0; s >>= 1) { if (t < s) sm[t] = fmaxf(sm[t], sm[t + s]); __syncthreads(); }
  float mx = sm[0]; __syncthreads();
  sm[t] = __expf(v - mx); __syncthreads();
  for (int s = 128; s > 0; s >>= 1) { if (t < s) sm[t] += sm[t + s]; __syncthreads(); }
  float lse = mx + __logf(sm[0]);
  la[base + (long)t * stride] = v - lse;
}

// plan = exp(logits): writes plan (b,q,c) f16 and plan^T (b,c,q) f16
__global__ void k_exp(const float* __restrict__ la, half_t* __restrict__ ph,
                      half_t* __restrict__ phT, long n) {
  long i = (long)blockIdx.x * blockDim.x + threadIdx.x;
  if (i >= n) return;
  float p = __expf(la[i]);
  ph[i] = (half_t)p;
  long b = i >> 16, q = (i >> 8) & 255, c = i & 255;
  phT[(b << 16) + (c << 8) + q] = (half_t)p;
}

__global__ void k_inter(float* __restrict__ store, const float* __restrict__ qfc,
                        const float* __restrict__ cfq, const int* __restrict__ edge_indices,
                        int E) {
  long i = (long)blockIdx.x * blockDim.x + threadIdx.x;
  if (i >= (long)E * 640) return;
  int e = (int)(i / 640), j = (int)(i % 640);
  int slot = edge_indices[e];
  int g = slot >> 8, pos = slot & 255;
  const float* src = (g & 1) ? cfq : qfc;
  store[(long)e * 768 + 128 + j] = src[((long)(g >> 1) * 256 + pos) * 640 + j];
}

__global__ void k_scores(float* __restrict__ out, const float* __restrict__ qfin,
                         const float* __restrict__ pc) {
  int b = blockIdx.x, t = threadIdx.x;
  float acc = 0.f;
  for (int i = t; i < 256 * 128; i += 256) {
    float d = qfin[(long)b * 32768 + i] - pc[(long)b * 32768 + i];
    acc += d > 0.f ? d : 0.f;
  }
  __shared__ float sm[256];
  sm[t] = acc; __syncthreads();
  for (int s = 128; s > 0; s >>= 1) { if (t < s) sm[t] += sm[t + s]; __syncthreads(); }
  if (t == 0) out[b] = -sm[0];
}

// ---------------------------------------------------------------------------
extern "C" void kernel_launch(void* const* d_in, const int* in_sizes, int n_in,
                              void* d_out, int out_size, void* d_ws, size_t ws_size,
                              hipStream_t stream) {
  (void)n_in; (void)out_size; (void)ws_size;
  const long Nn = in_sizes[0] / 32;   // nodes (5120), multiple of 32
  const long E  = in_sizes[18];       // edges (20288), multiple of 32

  const float* node_features = (const float*)d_in[0];
  const float* edge_features = (const float*)d_in[1];
  const float* enc_node_W = (const float*)d_in[2];  const float* enc_node_b = (const float*)d_in[3];
  const float* enc_edge_W = (const float*)d_in[4];  const float* enc_edge_b = (const float*)d_in[5];
  const float* msg_W1 = (const float*)d_in[6];      const float* msg_b1 = (const float*)d_in[7];
  const float* msg_W2 = (const float*)d_in[8];      const float* msg_b2 = (const float*)d_in[9];
  const float* upd_W1 = (const float*)d_in[10];     const float* upd_b1 = (const float*)d_in[11];
  const float* upd_W2 = (const float*)d_in[12];     const float* upd_b2 = (const float*)d_in[13];
  const float* t1_W = (const float*)d_in[14];       const float* t1_b = (const float*)d_in[15];
  const float* t2_W = (const float*)d_in[16];       const float* t2_b = (const float*)d_in[17];
  const int* from_idx = (const int*)d_in[18];
  const int* to_idx = (const int*)d_in[19];
  const int* edge_indices = (const int*)d_in[20];
  const int* q_sizes = (const int*)d_in[21];
  const int* c_sizes = (const int*)d_in[22];
  float* scores = (float*)d_out;

  // ---- workspace carve-out (all 256B aligned) ----
  char* wsp = (char*)d_ws;
  size_t off = 0;
  auto allocB = [&](size_t bytes) -> void* {
    void* p = wsp + off; off += (bytes + 255) & ~(size_t)255; return p;
  };
  auto allocH = [&](long n) { return (half_t*)allocB((size_t)n * sizeof(half_t)); };
  auto allocF = [&](long n) { return (float*)allocB((size_t)n * sizeof(float)); };

  // weights stored TRANSPOSED (N x K) for contiguous B-fragment loads
  half_t* w_encn = allocH(64 * 32);
  half_t* w_ence = allocH(128 * 32);      // K padded 16 -> 32
  half_t* w_m1   = allocH(256 * 256);
  half_t* w_m2   = allocH(128 * 256);
  half_t* w_u1   = allocH(128 * 192);
  half_t* w_u2   = allocH(64 * 128);
  half_t* w_t1   = allocH(64 * 128);
  half_t* w_t2   = allocH(64 * 64);
  half_t* nf16   = allocH(Nn * 32);
  half_t* ef16   = allocH(E * 32);        // K padded 16 -> 32
  float*  encN   = allocF(Nn * 64);
  half_t* encNh  = allocH(Nn * 64);
  float*  nodes  = allocF(Nn * 64);
  half_t* nodesh = allocH(Nn * 64);
  float*  store  = allocF(E * 768);       // [:,0:128]=enc_edges, [:,128:]=inter
  float*  msgst  = allocF(E * 640);       // messages p=1..5 at cols (p-1)*128
  half_t* m_in   = allocH(E * 256);
  half_t* hbuf   = allocH(E * 256);
  float*  agg    = allocF(Nn * 128);
  half_t* u_in   = allocH(Nn * 192);
  half_t* uh     = allocH(Nn * 128);
  half_t* flat   = allocH(128L * 256 * 640);
  float*  qfin   = allocF(64L * 256 * 128);
  float*  cfin   = allocF(64L * 256 * 128);
  half_t* thq    = allocH(64L * 256 * 64);
  half_t* thc    = allocH(64L * 256 * 64);
  half_t* tq     = allocH(64L * 256 * 64);
  half_t* tc     = allocH(64L * 256 * 64);
  float*  logits = allocF(64L * 256 * 256);
  half_t* plan16 = allocH(64L * 256 * 256);
  half_t* plnT16 = allocH(64L * 256 * 256);
  float*  qfc    = allocF(64L * 256 * 640);
  float*  cfq    = allocF(64L * 256 * 640);
  float*  tmpf   = allocF(64L * 256 * 128);

  auto cvtT = [&](half_t* d, const float* s, int Ncols, int Krows) {
    k_cvt_t<<<CDIV((long)Ncols * Krows, 256), 256, 0, stream>>>(d, s, Ncols, Krows);
  };
  auto zf32 = [&](float* d, long n) {   // n % 4 == 0
    k_zero_f32v<<<CDIV(n / 4, 256), 256, 0, stream>>>((v4f*)d, n / 4);
  };
  auto zf16 = [&](half_t* d, long n) {  // n % 8 == 0
    k_zero_f16v<<<CDIV(n / 8, 256), 256, 0, stream>>>((v8h*)d, n / 8);
  };
  auto gemm = [&](const half_t* A, const half_t* B, const float* bias,
                  float* Cf, half_t* Ch, int M, int N, int K,
                  int lda, int ldb, int ldc, long sA, long sB, long sC,
                  float scale, int act, int tb, int batch) {
    dim3 grid(CDIV(M, 32), CDIV(N, 256), batch);
    if (act) {
      if (tb) k_gemm<1,1><<<grid,128,0,stream>>>(A,B,bias,Cf,Ch,M,N,K,lda,ldb,ldc,sA,sB,sC,scale);
      else    k_gemm<1,0><<<grid,128,0,stream>>>(A,B,bias,Cf,Ch,M,N,K,lda,ldb,ldc,sA,sB,sC,scale);
    } else {
      if (tb) k_gemm<0,1><<<grid,128,0,stream>>>(A,B,bias,Cf,Ch,M,N,K,lda,ldb,ldc,sA,sB,sC,scale);
      else    k_gemm<0,0><<<grid,128,0,stream>>>(A,B,bias,Cf,Ch,M,N,K,lda,ldb,ldc,sA,sB,sC,scale);
    }
  };

  // ---- per-launch setup ----
  cvtT(w_encn, enc_node_W, 64, 32);
  zf16(w_ence, 128 * 32);
  k_cvt_ts<<<CDIV(128L * 16, 256), 256, 0, stream>>>(w_ence, enc_edge_W, 128, 16, 32);
  cvtT(w_m1, msg_W1, 256, 256);
  cvtT(w_m2, msg_W2, 128, 256);
  cvtT(w_u1, upd_W1, 128, 192);
  cvtT(w_u2, upd_W2, 64, 128);
  cvtT(w_t1, t1_W, 64, 128);
  cvtT(w_t2, t2_W, 64, 64);
  k_cvt<<<CDIV(Nn * 32, 256), 256, 0, stream>>>(nf16, node_features, Nn * 32);
  zf16(ef16, E * 32);
  k_cvt_pad<<<CDIV(E * 16, 256), 256, 0, stream>>>(ef16, edge_features, (int)E);
  zf32(store, E * 768);

  // encoders
  gemm(nf16, w_encn, enc_node_b, encN, encNh, (int)Nn, 64, 32, 32, 32, 64,
       0, 0, 0, 1.f, 0, 1, 1);
  gemm(ef16, w_ence, enc_edge_b, store, nullptr, (int)E, 128, 32, 32, 32, 768,
       0, 0, 0, 1.f, 0, 1, 1);

  const long FG = 256L * 640;
  for (int t = 0; t < 3; ++t) {
    k_copy_f32v<<<CDIV(Nn * 16, 256), 256, 0, stream>>>((v4f*)nodes, (const v4f*)encN, Nn * 16);
    k_copy_f16v<<<CDIV(Nn * 8, 256), 256, 0, stream>>>((v8h*)nodesh, (const v8h*)encNh, Nn * 8);

    for (int p = 1; p <= 5; ++p) {
      k_build_min<<<CDIV(E * 256, 256), 256, 0, stream>>>(
          m_in, nodesh, store + (long)(p - 1) * 128, 768, from_idx, to_idx, (int)E);
      gemm(m_in, w_m1, msg_b1, nullptr, hbuf, (int)E, 256, 256, 256, 256, 256,
           0, 0, 0, 1.f, 1, 1, 1);
      gemm(hbuf, w_m2, msg_b2, msgst + (long)(p - 1) * 128, nullptr,
           (int)E, 128, 256, 256, 256, 640, 0, 0, 0, 1.f, 0, 1, 1);
      zf32(agg, Nn * 128);
      k_segsum<<<CDIV(E * 128, 256), 256, 0, stream>>>(
          agg, msgst + (long)(p - 1) * 128, 640, to_idx, (int)E);
      k_build_uin<<<CDIV(Nn * 192, 256), 256, 0, stream>>>(u_in, nodesh, agg, (int)Nn);
      gemm(u_in, w_u1, upd_b1, nullptr, uh, (int)Nn, 128, 192, 192, 192, 128,
           0, 0, 0, 1.f, 1, 1, 1);
      gemm(uh, w_u2, upd_b2, nodes, nodesh, (int)Nn, 64, 128, 128, 128, 64,
           0, 0, 0, 1.f, 0, 1, 1);
    }

    zf16(flat, 128L * 256 * 640);
    zf32(qfin, 64L * 256 * 128);
    zf32(cfin, 64L * 256 * 128);
    k_scatter<<<CDIV(E * 640, 256), 256, 0, stream>>>(flat, qfin, cfin, msgst,
                                                      edge_indices, (int)E);

    gemm(flat + 512, w_t1, t1_b, nullptr, thq, 256, 64, 128, 640, 128, 64,
         2 * FG, 0, 256L * 64, 1.f, 1, 1, 64);
    gemm(flat + FG + 512, w_t1, t1_b, nullptr, thc, 256, 64, 128, 640, 128, 64,
         2 * FG, 0, 256L * 64, 1.f, 1, 1, 64);
    gemm(thq, w_t2, t2_b, nullptr, tq, 256, 64, 64, 64, 64, 64,
         256L * 64, 0, 256L * 64, 1.f, 0, 1, 64);
    gemm(thc, w_t2, t2_b, nullptr, tc, 256, 64, 64, 64, 64, 64,
         256L * 64, 0, 256L * 64, 1.f, 0, 1, 64);
    k_mask_rows<<<CDIV(64L * 256 * 64, 256), 256, 0, stream>>>(tq, q_sizes);
    k_mask_rows<<<CDIV(64L * 256 * 64, 256), 256, 0, stream>>>(tc, c_sizes);

    // logits = (tq @ tc^T) / SINK_TEMP   (tc natural layout is N x K -> TB=1)
    gemm(tq, tc, nullptr, logits, nullptr, 256, 256, 64, 64, 64, 256,
         256L * 64, 256L * 64, 65536L, 10.f, 0, 1, 64);

    for (int it = 0; it < 10; ++it) {
      k_sink_norm<<<64 * 256, 256, 0, stream>>>(logits, 0);
      k_sink_norm<<<64 * 256, 256, 0, stream>>>(logits, 1);
    }
    k_exp<<<CDIV(64L * 65536, 256), 256, 0, stream>>>(logits, plan16, plnT16,
                                                      64L * 65536);

    // plan @ c_store[:,:,128:]  and  plan^T @ q_store[:,:,128:]
    gemm(plan16, flat + FG, nullptr, qfc, nullptr, 256, 640, 256, 256, 640, 640,
         65536L, 2 * FG, 256L * 640, 1.f, 0, 0, 64);
    gemm(plnT16, flat, nullptr, cfq, nullptr, 256, 640, 256, 256, 640, 640,
         65536L, 2 * FG, 256L * 640, 1.f, 0, 0, 64);

    k_inter<<<CDIV(E * 640, 256), 256, 0, stream>>>(store, qfc, cfq, edge_indices, (int)E);
  }

  // scores = -sum relu(q_final - plan @ c_final)
  gemm(plan16, flat + FG + 512, nullptr, tmpf, nullptr, 256, 128, 256,
       256, 640, 128, 65536L, 2 * FG, 256L * 128, 1.f, 0, 0, 64);
  k_scores<<<64, 256, 0, stream>>>(scores, qfin, tmpf);
}
